// DynamicAdjacencyLayer_83141976916948
// MI455X (gfx1250) — compile-verified, matrix operand order checked by reading, not run on hardware
//
#include <hip/hip_runtime.h>
#include <hip/hip_bf16.h>

// MI455X / gfx1250, wave32. Gram+softmax with v_wmma_f32_16x16x32_bf16,
// B-panels staged in LDS by the Tensor Data Mover (tensor_load_to_lds).
typedef __attribute__((ext_vector_type(16))) __bf16 v16bf;
typedef __attribute__((ext_vector_type(8)))  float  v8f;
typedef __attribute__((ext_vector_type(4)))  unsigned int u32x4;
typedef __attribute__((ext_vector_type(8)))  int          i32x8;
typedef __attribute__((ext_vector_type(4)))  int          i32x4;

constexpr int kB = 8;     // batches
constexpr int kN = 2048;  // rows (= cols of Gram)
constexpr int kD = 256;   // feature dim (= K of GEMM)

// LDS B-panel staging geometry
constexpr int CHUNK_ROWS  = 64;                    // rows of u per TDM chunk (4 N-tiles)
constexpr int CHUNK_TILES = CHUNK_ROWS / 16;       // 4
constexpr int N_CHUNKS    = kN / CHUNK_ROWS;       // 32
constexpr int ROW_BYTES   = kD * 2;                // 512 B of bf16 per row
constexpr int ROW_PITCH   = ROW_BYTES + 16;        // +16B TDM pad -> bank-conflict-free
constexpr int CHUNK_BYTES = CHUNK_ROWS * ROW_PITCH;        // 33792 B per matrix
constexpr int BUF_BYTES   = 2 * CHUNK_BYTES;               // hi + lo
constexpr int LDS_DYN     = 2 * BUF_BYTES;                 // double buffered = 135168 B

union FragU { v16bf v; uint4 q[2]; };  // one 16x16x32 bf16 A/B fragment per lane

// ---------------------------------------------------------------------------
// TDM: async-copy a 64-row x 256-col bf16 tile (rows contiguous, stride 256)
// from global to LDS at lds_off, inserting 16B padding after each 512B row.
// D# packing per CDNA5 ISA.
// ---------------------------------------------------------------------------
__device__ __forceinline__ void tdm_load_panel(const void* gptr, unsigned lds_off) {
  unsigned long long ga = (unsigned long long)gptr;
  u32x4 g0;
  g0[0] = 1u;                                            // count=1 (valid), user D#
  g0[1] = lds_off;                                       // lds_addr [63:32]
  g0[2] = (unsigned)ga;                                  // global_addr [95:64]
  g0[3] = (unsigned)((ga >> 32) & 0x1FFFFFFu)            // global_addr [120:96]
        | (2u << 30);                                    // type=2 ("image")
  i32x8 g1;
  g1[0] = (1 << 16)                                      // data_size: 2 bytes
        | (1 << 20)                                      // pad_enable
        | (6 << 22)                                      // pad_interval: 128 dwords (512B)
        | (3 << 25);                                     // pad_amount: 4 dwords (16B)
  g1[1] = (int)(256u << 16);                             // tensor_dim0 = 256
  g1[2] = (int)((unsigned)CHUNK_ROWS << 16);             // tensor_dim1 = 64
  g1[3] = (int)(256u << 16);                             // tile_dim0 = 256
  g1[4] = CHUNK_ROWS;                                    // tile_dim1 = 64, tile_dim2 = 0
  g1[5] = 256;                                           // tensor_dim0_stride = 256
  g1[6] = 0;
  g1[7] = 0;
  i32x4 z4 = {0, 0, 0, 0};
#if __clang_major__ >= 23
  i32x8 z8 = {0, 0, 0, 0, 0, 0, 0, 0};
  __builtin_amdgcn_tensor_load_to_lds(g0, g1, z4, z4, z8, 0);
#else
  __builtin_amdgcn_tensor_load_to_lds(g0, g1, z4, z4, 0);
#endif
}

// ---------------------------------------------------------------------------
// Phase 1: row-normalize h (Gram of u == cosine similarity) and split each
// normalized row into bf16 hi + bf16 lo residual (3xWMMA split ~ fp32 accuracy).
// ---------------------------------------------------------------------------
__global__ void normalize_split_kernel(const float* __restrict__ h,
                                       __hip_bfloat16* __restrict__ uhi,
                                       __hip_bfloat16* __restrict__ ulo) {
  const int lane = threadIdx.x & 31;
  const int wave = threadIdx.x >> 5;
  const int row  = blockIdx.x * 8 + wave;            // 0 .. kB*kN-1

  const float* p = h + (size_t)row * kD + lane * 8;
  float x[8];
#pragma unroll
  for (int i = 0; i < 8; ++i) x[i] = p[i];

  float ss = 0.f;
#pragma unroll
  for (int i = 0; i < 8; ++i) ss += x[i] * x[i];
#pragma unroll
  for (int off = 16; off > 0; off >>= 1) ss += __shfl_xor(ss, off, 32);

  const float inv = rsqrtf(fmaxf(ss, 1e-16f));       // eps guard; ||h||~16 here

  union { __hip_bfloat16 b[8]; uint4 q; } hi, lo;
#pragma unroll
  for (int i = 0; i < 8; ++i) {
    float u = x[i] * inv;
    __hip_bfloat16 uh = __float2bfloat16(u);
    hi.b[i] = uh;
    lo.b[i] = __float2bfloat16(u - __bfloat162float(uh));
  }
  *(uint4*)(uhi + (size_t)row * kD + lane * 8) = hi.q;
  *(uint4*)(ulo + (size_t)row * kD + lane * 8) = lo.q;
}

// ---------------------------------------------------------------------------
// Phase 2: fused Gram (split-bf16 WMMA) + softmax.
// Grid: (16 strips, 8 batches), 256 threads = 8 waves.
// Workgroup owns 128 rows x all 2048 cols; wave w owns the 16-row M-subtile w.
// B-panels (64 rows, hi+lo) are TDM-staged into double-buffered LDS by wave 0
// and consumed by all 8 waves (ds_load_b128 fragments), overlapping DMA with
// WMMA via TENSORcnt. exp(sim/T) is written out while row sums accumulate in
// LDS; each lane then rescales exactly the addresses it stored (same-wave
// same-address store->load is kept in order).
// ---------------------------------------------------------------------------
__global__ __launch_bounds__(256)
void gram_softmax_kernel(const __hip_bfloat16* __restrict__ uhi,
                         const __hip_bfloat16* __restrict__ ulo,
                         const float* __restrict__ temp,
                         float* __restrict__ out) {
  extern __shared__ char smem[];                 // LDS_DYN bytes, TDM destination
  __shared__ float rowsum[128];

  const int lane = threadIdx.x & 31;
  const int wave = threadIdx.x >> 5;
  const int strip = blockIdx.x;                  // 0..15
  const int batch = blockIdx.y;                  // 0..7

  if (threadIdx.x < 128) rowsum[threadIdx.x] = 0.f;

  const float invT = 1.0f / temp[0];
  const int lg = lane >> 4;                      // K-half select within fragment
  const int lr = lane & 15;                      // M (A) / N (B) index within tile

  // ---- A fragments: this wave's 16 rows, all K, hi+lo, register-resident ----
  const int m0 = strip * 128 + wave * 16;
  const size_t arow = ((size_t)batch * kN + m0 + lr) * kD;
  const char* ahp = (const char*)(uhi + arow) + lg * 16;
  const char* alp = (const char*)(ulo + arow) + lg * 16;

  FragU ah[8], al[8];
#pragma unroll
  for (int ks = 0; ks < 8; ++ks) {
    ah[ks].q[0] = *(const uint4*)(ahp + ks * 64);
    ah[ks].q[1] = *(const uint4*)(ahp + ks * 64 + 32);
    al[ks].q[0] = *(const uint4*)(alp + ks * 64);
    al[ks].q[1] = *(const uint4*)(alp + ks * 64 + 32);
  }

  float psum[8];
#pragma unroll
  for (int v = 0; v < 8; ++v) psum[v] = 0.f;

  const int rbase = m0 + 8 * lg;                 // C layout: row = rbase + vgpr idx
  float* outb = out + (size_t)batch * kN * kN;

  const char* ub_hi = (const char*)(uhi + (size_t)batch * kN * kD);
  const char* ub_lo = (const char*)(ulo + (size_t)batch * kN * kD);
  const unsigned lds0 = (unsigned)(size_t)(void*)smem;   // low 32 bits = LDS offset

  // ---- prologue: kick off chunk 0 DMA ----
  if (wave == 0 && lane == 0) {
    tdm_load_panel(ub_hi, lds0);
    tdm_load_panel(ub_lo, lds0 + CHUNK_BYTES);
  }
  __syncthreads();   // also covers rowsum init

  for (int c = 0; c < N_CHUNKS; ++c) {
    // issue next chunk into the other buffer, then wait for this chunk's 2 DMAs
    if (wave == 0 && lane == 0) {
      if (c + 1 < N_CHUNKS) {
        const unsigned boff = lds0 + ((c + 1) & 1) * BUF_BYTES;
        tdm_load_panel(ub_hi + (size_t)(c + 1) * CHUNK_ROWS * ROW_BYTES, boff);
        tdm_load_panel(ub_lo + (size_t)(c + 1) * CHUNK_ROWS * ROW_BYTES, boff + CHUNK_BYTES);
        __builtin_amdgcn_s_wait_tensorcnt(2);    // chunk c done; c+1 in flight
      } else {
        __builtin_amdgcn_s_wait_tensorcnt(0);
      }
    }
    __syncthreads();                             // chunk c visible to all waves

    const char* lhi = smem + (c & 1) * BUF_BYTES;
    const char* llo = lhi + CHUNK_BYTES;

#pragma unroll
    for (int t = 0; t < CHUNK_TILES; ++t) {
      const int n0 = c * CHUNK_ROWS + t * 16;
      const char* bhp = lhi + (t * 16 + lr) * ROW_PITCH + lg * 16;
      const char* blp = llo + (t * 16 + lr) * ROW_PITCH + lg * 16;

      v8f acc = {};
#pragma unroll
      for (int ks = 0; ks < 8; ++ks) {
        FragU bh, bl;
        bh.q[0] = *(const uint4*)(bhp + ks * 64);
        bh.q[1] = *(const uint4*)(bhp + ks * 64 + 32);
        bl.q[0] = *(const uint4*)(blp + ks * 64);
        bl.q[1] = *(const uint4*)(blp + ks * 64 + 32);
        // u*u^T ~= hi*hi + hi*lo + lo*hi  (lo*lo ~ 2^-16, dropped)
        acc = __builtin_amdgcn_wmma_f32_16x16x32_bf16(false, ah[ks].v, false, bh.v,
                                                      (short)0, acc, false, false);
        acc = __builtin_amdgcn_wmma_f32_16x16x32_bf16(false, ah[ks].v, false, bl.v,
                                                      (short)0, acc, false, false);
        acc = __builtin_amdgcn_wmma_f32_16x16x32_bf16(false, al[ks].v, false, bh.v,
                                                      (short)0, acc, false, false);
      }

      // sim in [-1,1] -> exp without max-subtraction is exact & safe
      const int col = n0 + lr;
#pragma unroll
      for (int v = 0; v < 8; ++v) {
        float e = __expf(acc[v] * invT);
        psum[v] += e;
        outb[(size_t)(rbase + v) * kN + col] = e;
      }
    }
    __syncthreads();    // all waves done reading buf[c&1] before it is re-filled
  }

  // ---- per-row softmax sums (16 N-lanes per row) via LDS float atomics ----
  const int lrow = wave * 16 + 8 * lg;
#pragma unroll
  for (int v = 0; v < 8; ++v) atomicAdd(&rowsum[lrow + v], psum[v]);
  __syncthreads();

  float rinv[8];
#pragma unroll
  for (int v = 0; v < 8; ++v) rinv[v] = 1.0f / rowsum[lrow + v];

  // ---- rescale own stores (output strip is L2-resident: 134 MB < 192 MB) ----
  for (int nt = 0; nt < kN / 16; ++nt) {
    const int col = nt * 16 + lr;
#pragma unroll
    for (int v = 0; v < 8; ++v) {
      const size_t idx = (size_t)(rbase + v) * kN + col;
      outb[idx] *= rinv[v];
    }
  }
}

// ---------------------------------------------------------------------------
extern "C" void kernel_launch(void* const* d_in, const int* in_sizes, int n_in,
                              void* d_out, int out_size, void* d_ws, size_t ws_size,
                              hipStream_t stream) {
  const float* h    = (const float*)d_in[0];
  const float* temp = (const float*)d_in[1];   // scalar on device
  float* out        = (float*)d_out;

  // workspace: u_hi then u_lo, each [8][2048][256] bf16 (8 MB each)
  __hip_bfloat16* uhi = (__hip_bfloat16*)d_ws;
  __hip_bfloat16* ulo = uhi + (size_t)kB * kN * kD;

  normalize_split_kernel<<<dim3(kB * kN / 8), 256, 0, stream>>>(h, uhi, ulo);
  gram_softmax_kernel<<<dim3(kN / 128, kB), 256, LDS_DYN, stream>>>(uhi, ulo, temp, out);
}
